// RNN_64269890618014
// MI455X (gfx1250) — compile-verified, matrix operand order checked by reading, not run on hardware
//
#include <hip/hip_runtime.h>
#include <hip/hip_bf16.h>

typedef _Float16 v16h __attribute__((ext_vector_type(16)));
typedef _Float16 h8   __attribute__((ext_vector_type(8)));
typedef _Float16 h4   __attribute__((ext_vector_type(4)));
typedef float    v8f  __attribute__((ext_vector_type(8)));
typedef float    f4   __attribute__((ext_vector_type(4)));

#define T_STEPS 2048
#define BATCHSZ 64
#define EMBED   256
#define HIDDEN  64
#define GATES   256   // 4*HIDDEN

// ---- helpers -------------------------------------------------------------

__device__ __forceinline__ h8 cvt8(f4 a, f4 b) {
    h8 r = {(_Float16)a[0], (_Float16)a[1], (_Float16)a[2], (_Float16)a[3],
            (_Float16)b[0], (_Float16)b[1], (_Float16)b[2], (_Float16)b[3]};
    return r;
}

__device__ __forceinline__ v16h pack16(f4 a, f4 b, f4 c, f4 d) {
    return __builtin_shufflevector(cvt8(a, b), cvt8(c, d),
                                   0,1,2,3,4,5,6,7,8,9,10,11,12,13,14,15);
}

__device__ __forceinline__ v16h join16(h8 lo, h8 hi) {
    return __builtin_shufflevector(lo, hi,
                                   0,1,2,3,4,5,6,7,8,9,10,11,12,13,14,15);
}

__device__ __forceinline__ float sigmoidf_(float x) {
    return 1.0f / (1.0f + __expf(-x));
}

__device__ __forceinline__ float tanhf_(float x) {
    x = fminf(15.0f, fmaxf(-15.0f, x));
    float e = __expf(2.0f * x);
    return (e - 1.0f) / (e + 1.0f);
}

// ---- Kernel 0: pre-pack W_ih into WMMA B-fragment layout (f16) -----------
// 128 fragments = 16 n-tiles x 8 k-chunks; 32 lanes x 16 halves each.
// wsB[frag*512 + lane*16 .. +16) = B-fragment halves for (frag, lane).

__global__ __launch_bounds__(256)
void pack_wih_kernel(const float* __restrict__ W_ih, _Float16* __restrict__ wsB)
{
    int gid   = blockIdx.x * 256 + threadIdx.x;   // 0..4095
    int lane  = gid & 31;
    int frag  = gid >> 5;                         // 0..127
    int kc    = frag >> 4;
    int nt    = frag & 15;
    int l16   = lane & 15;
    int lhalf = lane >> 4;
    int n  = nt * 16 + l16;
    int kb = kc * 32 + lhalf * 8;                 // K halves {kb..+7, kb+16..+23}
    const f4* p = (const f4*)(W_ih + (size_t)n * EMBED + kb);
    v16h v = pack16(p[0], p[1], p[4], p[5]);
    *(v16h*)(wsB + (size_t)frag * 512 + lane * 16) = v;
}

// ---- Kernel 1: gx[t,b,:] = W_ih @ emb[x[b,t]] + b_ih + b_hh --------------
// GEMM: M = T*B (tiles of 16 rows), N = 256 (gates), K = 256 (embed).
// One workgroup = one M-tile (16 rows, all same t). 8 waves cover N=256.

__global__ __launch_bounds__(256)
void lstm_gx_kernel(const int* __restrict__ x, const float* __restrict__ emb,
                    const _Float16* __restrict__ wsB, const float* __restrict__ b_ih,
                    const float* __restrict__ b_hh, float* __restrict__ gx)
{
    __shared__ _Float16 aS[16 * 264];   // 16 rows, stride 264 halves (528 B)

    const int tid = threadIdx.x;
    const int t   = blockIdx.x >> 2;           // 4 tiles of 16 batch rows per t
    const int b0  = (blockIdx.x & 3) << 4;

    // Stage A: gather 16 embedding rows, convert to f16 into LDS.
    {
        int r = tid >> 4, s = tid & 15;
        int b = b0 + r;
        int xi = x[(size_t)b * T_STEPS + t];
        const f4* src = (const f4*)(emb + (size_t)xi * EMBED + s * 16);
        f4 p0 = src[0], p1 = src[1], p2 = src[2], p3 = src[3];
        h8* dst = (h8*)(aS + r * 264 + s * 16);
        dst[0] = cvt8(p0, p1);
        dst[1] = cvt8(p2, p3);
    }
    __syncthreads();

    const int wave  = tid >> 5;
    const int lane  = tid & 31;
    const int l16   = lane & 15;
    const int lhalf = (lane < 16) ? 0 : 1;

    v8f c0 = {0,0,0,0,0,0,0,0};
    v8f c1 = {0,0,0,0,0,0,0,0};
    const int n0 = wave * 32 + l16;
    const int n1 = n0 + 16;

    #pragma unroll
    for (int kc = 0; kc < 8; ++kc) {
        // A fragment from LDS: lane holds M=l16, K halves {base..+7, base+16..+23}
        const _Float16* ar = aS + l16 * 264 + kc * 32 + lhalf * 8;
        v16h a = join16(*(const h8*)ar, *(const h8*)(ar + 16));
        // Pre-packed B fragments: direct f16 coalesced loads, no conversion.
        const _Float16* bp0 = wsB + (size_t)(kc * 16 + 2 * wave) * 512 + lane * 16;
        v16h bf0 = join16(*(const h8*)bp0, *(const h8*)(bp0 + 8));
        const _Float16* bp1 = bp0 + 512;
        v16h bf1 = join16(*(const h8*)bp1, *(const h8*)(bp1 + 8));
        c0 = __builtin_amdgcn_wmma_f32_16x16x32_f16(false, a, false, bf0,
                                                    (short)0, c0, false, false);
        c1 = __builtin_amdgcn_wmma_f32_16x16x32_f16(false, a, false, bf1,
                                                    (short)0, c1, false, false);
    }

    float bias0 = b_ih[n0] + b_hh[n0];
    float bias1 = b_ih[n1] + b_hh[n1];
    size_t mbase = (size_t)blockIdx.x * 16;
    #pragma unroll
    for (int r = 0; r < 8; ++r) {
        size_t m = mbase + r + lhalf * 8;   // C layout: vgpr r -> row r / r+8
        gx[m * GATES + n0] = c0[r] + bias0;
        gx[m * GATES + n1] = c1[r] + bias1;
    }
}

// ---- Kernel 2: sequential LSTM scan + BN + FC ----------------------------
// 4 workgroups, each owns 16 batch rows. Per step: gates = gx[t] + h @ W_hh^T
// via WMMA (M=16, N=256, K=64), then gate nonlinearities + state update.

__global__ __launch_bounds__(256)
void lstm_scan_kernel(const float* __restrict__ gx, const float* __restrict__ W_hh,
                      const float* __restrict__ bn_gamma, const float* __restrict__ bn_beta,
                      const float* __restrict__ bn_mean, const float* __restrict__ bn_var,
                      const float* __restrict__ fc_w, const float* __restrict__ fc_b,
                      float* __restrict__ out)
{
    __shared__ float    gatesS[16 * 260];   // padded stride to dodge bank conflicts
    __shared__ _Float16 hS[16 * 64];        // h state, f16, row stride 64 halves

    const int tid   = threadIdx.x;
    const int wg    = blockIdx.x;           // batch rows [wg*16, wg*16+16)
    const int wave  = tid >> 5;
    const int lane  = tid & 31;
    const int l16   = lane & 15;
    const int lhalf = (lane < 16) ? 0 : 1;

    // Preload W_hh as WMMA B fragments (resident for entire scan).
    v16h Bf[2][2];
    #pragma unroll
    for (int nt = 0; nt < 2; ++nt)
        #pragma unroll
        for (int kc = 0; kc < 2; ++kc) {
            int n  = wave * 32 + nt * 16 + l16;
            int kb = kc * 32 + lhalf * 8;
            const f4* p = (const f4*)(W_hh + (size_t)n * HIDDEN + kb);
            Bf[nt][kc] = pack16(p[0], p[1], p[4], p[5]);
        }

    for (int i = tid; i < 16 * 64; i += 256) hS[i] = (_Float16)0.0f;

    const int bl = tid >> 4;           // local batch row this thread owns
    const int j0 = (tid & 15) << 2;    // 4 hidden units per thread
    float cr[4] = {0.f, 0.f, 0.f, 0.f};
    float hr[4] = {0.f, 0.f, 0.f, 0.f};
    const float* gxb = gx + (size_t)(wg * 16 + bl) * GATES;
    __syncthreads();

    for (int t = 0; t < T_STEPS; ++t) {
        // Preload gx[t] for this thread's (b, j) slots — independent of h.
        const float* g = gxb + (size_t)t * BATCHSZ * GATES;
        f4 gi = *(const f4*)(g + 0 * HIDDEN + j0);
        f4 gf = *(const f4*)(g + 1 * HIDDEN + j0);
        f4 gg = *(const f4*)(g + 2 * HIDDEN + j0);
        f4 go = *(const f4*)(g + 3 * HIDDEN + j0);
        if (t + 1 < T_STEPS) {
            const float* gn = g + (size_t)BATCHSZ * GATES;
            __builtin_prefetch(gn + j0, 0, 0);
            __builtin_prefetch(gn + 64 + j0, 0, 0);
            __builtin_prefetch(gn + 128 + j0, 0, 0);
            __builtin_prefetch(gn + 192 + j0, 0, 0);
        }

        // WMMA phase: gates(M=16,N=256) = h(16x64) @ W_hh^T(64x256)
        v8f c0 = {0,0,0,0,0,0,0,0}, c1 = {0,0,0,0,0,0,0,0};
        {
            const _Float16* row = hS + l16 * 64;
            v16h a0 = join16(*(const h8*)(row + lhalf * 8),
                             *(const h8*)(row + lhalf * 8 + 16));
            v16h a1 = join16(*(const h8*)(row + 32 + lhalf * 8),
                             *(const h8*)(row + 32 + lhalf * 8 + 16));
            c0 = __builtin_amdgcn_wmma_f32_16x16x32_f16(false, a0, false, Bf[0][0],
                                                        (short)0, c0, false, false);
            c0 = __builtin_amdgcn_wmma_f32_16x16x32_f16(false, a1, false, Bf[0][1],
                                                        (short)0, c0, false, false);
            c1 = __builtin_amdgcn_wmma_f32_16x16x32_f16(false, a0, false, Bf[1][0],
                                                        (short)0, c1, false, false);
            c1 = __builtin_amdgcn_wmma_f32_16x16x32_f16(false, a1, false, Bf[1][1],
                                                        (short)0, c1, false, false);
        }
        {
            int n0 = wave * 32 + l16;
            #pragma unroll
            for (int r = 0; r < 8; ++r) {
                int m = r + lhalf * 8;
                gatesS[m * 260 + n0]      = c0[r];
                gatesS[m * 260 + n0 + 16] = c1[r];
            }
        }
        __syncthreads();

        // Elementwise phase: nonlinearity + state update, h back to LDS as f16.
        {
            const float* gr = gatesS + bl * 260;
            h4 hh;
            #pragma unroll
            for (int e = 0; e < 4; ++e) {
                int j = j0 + e;
                float iv = sigmoidf_(gi[e] + gr[j]);
                float fv = sigmoidf_(gf[e] + gr[64 + j]);
                float gv = tanhf_(gg[e] + gr[128 + j]);
                float ov = sigmoidf_(go[e] + gr[192 + j]);
                cr[e] = fv * cr[e] + iv * gv;
                float hv = ov * tanhf_(cr[e]);
                hr[e] = hv;
                hh[e] = (_Float16)hv;
            }
            *(h4*)(hS + bl * 64 + j0) = hh;
        }
        __syncthreads();
    }

    // BatchNorm (eval) into LDS (reuse gatesS as [16][64] f32).
    #pragma unroll
    for (int e = 0; e < 4; ++e) {
        int j = j0 + e;
        float hb = (hr[e] - bn_mean[j]) * rsqrtf(bn_var[j] + 1e-5f)
                   * bn_gamma[j] + bn_beta[j];
        gatesS[bl * 64 + j] = hb;
    }
    __syncthreads();

    // FC: out[b, q] = h_bn[b] . fc_w[q] + fc_b[q]; 16 b x 16 q = 256 threads.
    {
        int q = tid & 15;
        const float* hv = gatesS + bl * 64;
        const float* w  = fc_w + q * 64;
        float acc = fc_b[q];
        #pragma unroll
        for (int j = 0; j < 64; ++j) acc += hv[j] * w[j];
        out[(size_t)(wg * 16 + bl) * 16 + q] = acc;
    }
}

// ---- launch --------------------------------------------------------------

extern "C" void kernel_launch(void* const* d_in, const int* in_sizes, int n_in,
                              void* d_out, int out_size, void* d_ws, size_t ws_size,
                              hipStream_t stream)
{
    const int*   x     = (const int*)d_in[0];
    // d_in[1] = seq_lengths: unused by the reference computation
    const float* emb   = (const float*)d_in[2];
    const float* W_ih  = (const float*)d_in[3];
    const float* W_hh  = (const float*)d_in[4];
    const float* b_ih  = (const float*)d_in[5];
    const float* b_hh  = (const float*)d_in[6];
    const float* gamma = (const float*)d_in[7];
    const float* beta  = (const float*)d_in[8];
    const float* mean  = (const float*)d_in[9];
    const float* var   = (const float*)d_in[10];
    const float* fcw   = (const float*)d_in[11];
    const float* fcb   = (const float*)d_in[12];
    float* out = (float*)d_out;

    float*     gx  = (float*)d_ws;                       // 128 MB: T*B*4H f32
    _Float16*  wsB = (_Float16*)(gx + (size_t)T_STEPS * BATCHSZ * GATES); // +128 KB

    // Phase 0: one-time pre-pack of W_ih into WMMA B-fragment layout (f16).
    pack_wih_kernel<<<dim3(16), dim3(256), 0, stream>>>(W_ih, wsB);

    // Phase 1: embedding gather + input projection (BW-bound, parallel).
    lstm_gx_kernel<<<dim3(T_STEPS * 4), dim3(256), 0, stream>>>(
        x, emb, wsB, b_ih, b_hh, gx);

    // Phase 2: sequential scan (latency-bound) + BN + FC epilogue.
    lstm_scan_kernel<<<dim3(4), dim3(256), 0, stream>>>(
        gx, W_hh, gamma, beta, mean, var, fcw, fcb, out);
}
